// SCB_43026982371893
// MI455X (gfx1250) — compile-verified, matrix operand order checked by reading, not run on hardware
//
#include <hip/hip_runtime.h>
#include <hip/hip_bf16.h>
#include <math.h>

// ---------------------------------------------------------------------------
// SCB block for MI455X (gfx1250, wave32, WMMA).
// Pipeline: prep (S4D kernel + bf16 weights + padded bias) -> stage u ->
// fused S4D-conv + GELU + Wout GEMM (v_wmma_f32_16x16x32_bf16, async-LDS
// staging) + GLU + residual -> LN stats -> fused (Wg,Wi) 1x1-conv GEMMs +
// sigmoid gate -> LN stats -> 3-tap F-conv GEMM.
// ---------------------------------------------------------------------------

typedef __attribute__((ext_vector_type(16))) __bf16 v16bf;
typedef __attribute__((ext_vector_type(8)))  __bf16 v8bf;
typedef __attribute__((ext_vector_type(8)))  float  v8f;

#define NB   8
#define NC   64
#define NF   161      // = H (d_model)
#define NT   400
#define NL   64       // sequence length for S4D == channel count
#define NBT  (NB*NT)          // 3200
#define NCOL (NF*NT)          // 64400 columns (f,t) per batch
#define O2   322              // 2*H
#define OPAD 336              // 322 padded to 16
#define KPAD 192              // 161 padded to 32

// WMMA fragment loader: per-lane k indices for 16-bit A/B operands are
// k = 16*(i>>3) + 8*(lane>>4) + (i&7)  -> two contiguous 8-half (16B) groups.
__device__ __forceinline__ v16bf load_frag(const __bf16* p) {
    v8bf lo = *(const v8bf*)(p);
    v8bf hi = *(const v8bf*)(p + 16);
    v16bf r;
#pragma unroll
    for (int i = 0; i < 8; ++i) { r[i] = lo[i]; r[i + 8] = hi[i]; }
    return r;
}

__device__ __forceinline__ v8f wmma_bf16(v16bf a, v16bf b, v8f c) {
    return __builtin_amdgcn_wmma_f32_16x16x32_bf16(false, a, false, b,
                                                   (short)0, c, false, false);
}

__device__ __forceinline__ float sigmoidf_(float x) {
    return 1.0f / (1.0f + expf(-x));
}

// gfx1250 async global->LDS copies (ASYNCcnt-tracked, no VGPR round trip).
// LDS destination is the low 32 bits of the generic pointer (offset into the
// workgroup's LDS allocation), global source is a 64-bit VGPR pair.
__device__ __forceinline__ void async_load_b128(unsigned lds_off, const void* g) {
    asm volatile("global_load_async_to_lds_b128 %0, %1, off"
                 :: "v"(lds_off), "v"(g) : "memory");
}
__device__ __forceinline__ void async_load_b32(unsigned lds_off, const void* g) {
    asm volatile("global_load_async_to_lds_b32 %0, %1, off"
                 :: "v"(lds_off), "v"(g) : "memory");
}
__device__ __forceinline__ void wait_asynccnt0() {
    asm volatile("s_wait_asynccnt 0x0" ::: "memory");
}

// ---------------------------------------------------------------------------
// Kernel 1: precompute S4D conv kernel K[161][64], bf16 weights, padded bout.
// ---------------------------------------------------------------------------
#define PREP_NK   (NF*NL)              // 10304
#define PREP_NWO  (OPAD*KPAD)          // 64512
#define PREP_NWG  (NC*NC)              // 4096
#define PREP_NWC  (3*NC*NC)            // 12288
#define PREP_TOT  (PREP_NK + PREP_NWO + 2*PREP_NWG + PREP_NWC + OPAD)

__global__ void prep_kernel(const float* __restrict__ log_dt,
                            const float* __restrict__ log_A_real,
                            const float* __restrict__ A_imag,
                            const float* __restrict__ C_re,
                            const float* __restrict__ C_im,
                            const float* __restrict__ Wout,
                            const float* __restrict__ Wg,
                            const float* __restrict__ Wi,
                            const float* __restrict__ Wc,
                            const float* __restrict__ bout,
                            float* __restrict__ kf,
                            __bf16* __restrict__ woutbf,
                            __bf16* __restrict__ wgbf,
                            __bf16* __restrict__ wibf,
                            __bf16* __restrict__ wcbf,
                            float* __restrict__ boutp) {
    int idx = blockIdx.x * 256 + threadIdx.x;
    if (idx < PREP_NK) {
        int h = idx >> 6, l = idx & 63;
        float dt = expf(log_dt[h]);
        float acc = 0.0f;
        for (int n = 0; n < 32; ++n) {
            float Are = -expf(log_A_real[h * 32 + n]);
            float Aim = A_imag[h * 32 + n];
            float dr = Are * dt, di = Aim * dt;
            float ed = expf(dr);
            float er = ed * cosf(di) - 1.0f;        // exp(dtA) - 1
            float ei = ed * sinf(di);
            float den = Are * Are + Aim * Aim;      // |A|^2 (never 0 here)
            float qr = (er * Are + ei * Aim) / den; // (exp(dtA)-1)/A
            float qi = (ei * Are - er * Aim) / den;
            float cr = C_re[h * 32 + n], ci = C_im[h * 32 + n];
            float br = cr * qr - ci * qi;           // Cbar
            float bi = cr * qi + ci * qr;
            float el = expf(dr * (float)l);
            float ph = di * (float)l;
            float vr = el * cosf(ph), vi = el * sinf(ph); // exp(dtA*l)
            acc += br * vr - bi * vi;
        }
        kf[idx] = 2.0f * acc;
    } else if (idx < PREP_NK + PREP_NWO) {
        int r2 = idx - PREP_NK;
        int r = r2 / KPAD, k = r2 - r * KPAD;
        float v = (r < O2 && k < NF) ? Wout[r * NF + k] : 0.0f;
        woutbf[r2] = (__bf16)v;
    } else if (idx < PREP_NK + PREP_NWO + PREP_NWG) {
        int r2 = idx - (PREP_NK + PREP_NWO);
        wgbf[r2] = (__bf16)Wg[r2];
    } else if (idx < PREP_NK + PREP_NWO + 2 * PREP_NWG) {
        int r2 = idx - (PREP_NK + PREP_NWO + PREP_NWG);
        wibf[r2] = (__bf16)Wi[r2];
    } else if (idx < PREP_NK + PREP_NWO + 2 * PREP_NWG + PREP_NWC) {
        int r2 = idx - (PREP_NK + PREP_NWO + 2 * PREP_NWG);
        int df = r2 >> 12, r3 = r2 & 4095;
        int o = r3 >> 6, i = r3 & 63;
        wcbf[r2] = (__bf16)Wc[(o * 64 + i) * 3 + df];   // [df][o][i]
    } else if (idx < PREP_TOT) {
        int o = idx - (PREP_NK + PREP_NWO + 2 * PREP_NWG + PREP_NWC);
        boutp[o] = (o < O2) ? bout[o] : 0.0f;
    }
}

// ---------------------------------------------------------------------------
// Kernel 2: stage x (B,C,F,T) -> u_bf [bt][f][c] bf16 (coalesced over c).
// ---------------------------------------------------------------------------
__global__ void stage_u_kernel(const float* __restrict__ x,
                               __bf16* __restrict__ ubf) {
    int blk = blockIdx.x;
    int b = blk / (NF * 4);
    int rem = blk - b * NF * 4;
    int f = rem >> 2;
    int t0 = (rem & 3) * 100;
    for (int e = threadIdx.x; e < 64 * 100; e += 256) {
        int c = e & 63, t = t0 + (e >> 6);
        float v = x[(((size_t)(b * NC + c)) * NF + f) * NT + t];
        ubf[((size_t)(b * NT + t) * NF + f) * NL + c] = (__bf16)v;
    }
}

// ---------------------------------------------------------------------------
// Kernel 3: fused S4D causal conv + skip + GELU + Wout GEMM (WMMA bf16)
//           + bias + GLU + residual.  One workgroup (8 waves) per bt column.
// Dynamic LDS map:  [0..86016)  z   f32 [336][64]   (phase 3/4)
//                   [0..20608)  u   bf16 [161][64]  (phase 1/2, dies before z)
//                   [86016..110592) yT bf16 [64][192]
// ---------------------------------------------------------------------------
__global__ void __launch_bounds__(256)
s4d_glu_kernel(const __bf16* __restrict__ ubf,
               const float* __restrict__ kf,
               const float* __restrict__ Dskip,
               const __bf16* __restrict__ woutbf,
               const float* __restrict__ boutp,
               const float* __restrict__ x,
               float* __restrict__ xr) {
    extern __shared__ char smem[];
    __bf16* u  = (__bf16*)smem;                 // [161][64]
    float*  z  = (float*)smem;                  // [336][64]
    __bf16* yT = (__bf16*)(smem + OPAD * NL * 4); // [64][192]

    int bt = blockIdx.x;
    int lane = threadIdx.x & 31;
    int wave = threadIdx.x >> 5;

    // Phase 1: u tile -> LDS via gfx1250 async global->LDS b128 copies.
    // 20608 B = 40 * (32 lanes * 16 B) + 32 lanes * 4 B tail.
    {
        const char* gsrc = (const char*)(ubf + (size_t)bt * (NF * NL));
        unsigned ubase = (unsigned)(uintptr_t)(void*)u;
#pragma unroll
        for (int i = 0; i < 5; ++i) {
            unsigned off = (unsigned)(((wave * 5 + i) * 32 + lane) * 16);
            async_load_b128(ubase + off, gsrc + off);
        }
        if (wave == 0) {
            unsigned off = (unsigned)(40 * 512 + lane * 4);
            async_load_b32(ubase + off, gsrc + off);
        }
        wait_asynccnt0();
    }
    __syncthreads();

    // Phase 2: causal conv + Dskip + exact GELU -> yT[c][f] (bf16, f padded).
    for (int e = threadIdx.x; e < NL * KPAD; e += 256) {
        int c = e / KPAD, f = e - c * KPAD;
        __bf16 out;
        if (f < NF) {
            const __bf16* urow = u + f * NL;
            const float*  krow = kf + f * NL;
            float s = Dskip[f] * (float)urow[c];
            for (int j = 0; j <= c; ++j) s += krow[j] * (float)urow[c - j];
            float gl = 0.5f * s * (1.0f + erff(s * 0.70710678118654752f));
            out = (__bf16)gl;
        } else {
            out = (__bf16)0.0f;
        }
        yT[c * KPAD + f] = out;
    }
    __syncthreads();   // u is dead from here; z may alias it.

    // Phase 3: z[o][c] = Wout @ y  via 16x16x32 bf16 WMMA.
    {
        int g    = lane >> 4;
        int ln16 = lane & 15;
        int nt   = wave & 3;      // 4 N tiles (c)
        int half = wave >> 2;     // 2 wave-groups over 21 M tiles

        v16bf bfr[6];
        const __bf16* brow = yT + (nt * 16 + ln16) * KPAD;
#pragma unroll
        for (int kc = 0; kc < 6; ++kc)
            bfr[kc] = load_frag(brow + kc * 32 + g * 8);

        for (int mt = half; mt < OPAD / 16; mt += 2) {
            const __bf16* arow = woutbf + (mt * 16 + ln16) * KPAD;
            v8f acc = {0.f, 0.f, 0.f, 0.f, 0.f, 0.f, 0.f, 0.f};
#pragma unroll
            for (int kc = 0; kc < 6; ++kc)
                acc = wmma_bf16(load_frag(arow + kc * 32 + g * 8), bfr[kc], acc);
            int o0 = mt * 16 + g * 8;
            int cc = nt * 16 + ln16;
#pragma unroll
            for (int j = 0; j < 8; ++j)
                z[(o0 + j) * NL + cc] = acc[j] + boutp[o0 + j];  // unconditional
        }
    }
    __syncthreads();

    // Phase 4: GLU over channel halves + residual into xr (x layout).
    {
        int b = bt / NT, t = bt - (bt / NT) * NT;
        for (int e = threadIdx.x; e < NF * NL; e += 256) {
            int f = e >> 6, c = e & 63;
            float a  = z[f * NL + c];
            float gt = z[(f + NF) * NL + c];
            float v  = a * sigmoidf_(gt);
            size_t xi = (((size_t)(b * NC + c)) * NF + f) * NT + t;
            xr[xi] = v + x[xi];
        }
    }
}

// ---------------------------------------------------------------------------
// Kernel 4/6: per-(b,t) mean / 1/(std+1e-8) over (C,F), unbiased variance.
// ---------------------------------------------------------------------------
__global__ void ln_stats_kernel(const float* __restrict__ buf,
                                float* __restrict__ stats) {
    int bt = blockIdx.x;
    int b = bt / NT, t = bt - b * NT;
    size_t base = (size_t)b * NC * NCOL + t;
    float s = 0.0f, ss = 0.0f;
    for (int e = threadIdx.x; e < NC * NF; e += blockDim.x) {
        int c = e / NF, f = e - c * NF;
        float v = buf[base + ((size_t)c * NF + f) * NT];
        s += v; ss += v * v;
    }
#pragma unroll
    for (int o = 16; o > 0; o >>= 1) {
        s  += __shfl_down(s, o, 32);
        ss += __shfl_down(ss, o, 32);
    }
    __shared__ float ps[8], pss[8];
    int wid = threadIdx.x >> 5, ln = threadIdx.x & 31;
    if (ln == 0) { ps[wid] = s; pss[wid] = ss; }
    __syncthreads();
    if (threadIdx.x == 0) {
        float S = 0.f, SS = 0.f;
        for (int i = 0; i < 8; ++i) { S += ps[i]; SS += pss[i]; }
        const float n = (float)(NC * NF);
        float mean = S / n;
        float var  = (SS - n * mean * mean) / (n - 1.0f);
        var = var < 0.f ? 0.f : var;
        stats[bt * 2]     = mean;
        stats[bt * 2 + 1] = 1.0f / (sqrtf(var) + 1e-8f);
    }
}

// ---------------------------------------------------------------------------
// Kernel 5: h = (Wi@x + bi) * sigmoid(Wg@LN0(x) + bg)   (two fused WMMA GEMMs
// over the channel dim; B columns are (f,t) flattened, 64 per workgroup).
// ---------------------------------------------------------------------------
__global__ void __launch_bounds__(256)
gate_kernel(const float* __restrict__ xr, const float* __restrict__ st0,
            const float* __restrict__ w0, const float* __restrict__ b0,
            const __bf16* __restrict__ wgbf, const float* __restrict__ bg,
            const __bf16* __restrict__ wibf, const float* __restrict__ bi,
            float* __restrict__ hbuf) {
    __shared__ __bf16 Bn[64 * 72];   // [colLocal][c]  (LN0-normalized)
    __shared__ __bf16 Bx[64 * 72];   // [colLocal][c]  (raw)
    int b = blockIdx.y;
    int col0 = blockIdx.x * 64;

    for (int e = threadIdx.x; e < 64 * 64; e += 256) {
        int c = e >> 6, cl = e & 63;
        int col = col0 + cl;
        float v = 0.f, vn = 0.f;
        if (col < NCOL) {
            v = xr[((size_t)(b * NC + c)) * NCOL + col];
            int f = col / NT, t = col - f * NT;
            int bt = b * NT + t;
            float mean = st0[bt * 2], rstd = st0[bt * 2 + 1];
            vn = (v - mean) * rstd * w0[c * NF + f] + b0[c * NF + f];
        }
        Bx[cl * 72 + c] = (__bf16)v;
        Bn[cl * 72 + c] = (__bf16)vn;
    }
    __syncthreads();

    int lane = threadIdx.x & 31, wave = threadIdx.x >> 5;
    int g = lane >> 4, ln16 = lane & 15;
    for (int tt = wave; tt < 16; tt += 8) {
        int mt = tt >> 2, nt = tt & 3;
        const __bf16* brn = Bn + (nt * 16 + ln16) * 72;
        const __bf16* brx = Bx + (nt * 16 + ln16) * 72;
        const __bf16* arg = wgbf + (mt * 16 + ln16) * 64;
        const __bf16* ari = wibf + (mt * 16 + ln16) * 64;
        v8f ag = {0.f, 0.f, 0.f, 0.f, 0.f, 0.f, 0.f, 0.f};
        v8f ai = {0.f, 0.f, 0.f, 0.f, 0.f, 0.f, 0.f, 0.f};
#pragma unroll
        for (int kc = 0; kc < 2; ++kc) {
            ag = wmma_bf16(load_frag(arg + kc * 32 + g * 8),
                           load_frag(brn + kc * 32 + g * 8), ag);
            ai = wmma_bf16(load_frag(ari + kc * 32 + g * 8),
                           load_frag(brx + kc * 32 + g * 8), ai);
        }
#pragma unroll
        for (int j = 0; j < 8; ++j) {
            int o = mt * 16 + g * 8 + j;
            int col = col0 + nt * 16 + ln16;
            if (col < NCOL) {
                float zg = ag[j] + bg[o];
                float zi = ai[j] + bi[o];
                hbuf[((size_t)(b * NC + o)) * NCOL + col] = zi * sigmoidf_(zg);
            }
        }
    }
}

// ---------------------------------------------------------------------------
// Kernel 7: out = Wc(3-tap over F) @ LN1(h) + bc   (3 accumulated WMMA taps).
// ---------------------------------------------------------------------------
__global__ void __launch_bounds__(256)
outconv_kernel(const float* __restrict__ hbuf, const float* __restrict__ st1,
               const float* __restrict__ w1, const float* __restrict__ b1,
               const __bf16* __restrict__ wcbf, const float* __restrict__ bc,
               float* __restrict__ out) {
    __shared__ __bf16 Bt[3 * 64 * 72];   // per tap: [colLocal][c]
    int b = blockIdx.y;
    int col0 = blockIdx.x * 64;

    for (int e = threadIdx.x; e < 3 * 64 * 64; e += 256) {
        int df = e >> 12, r = e & 4095;
        int c = r >> 6, cl = r & 63;
        int col = col0 + cl;
        float vn = 0.f;
        if (col < NCOL) {
            int f = col / NT, t = col - f * NT;
            int fp = f + df - 1;
            if (fp >= 0 && fp < NF) {
                float v = hbuf[((size_t)(b * NC + c)) * NCOL + (size_t)fp * NT + t];
                int bt = b * NT + t;
                vn = (v - st1[bt * 2]) * st1[bt * 2 + 1] * w1[c * NF + fp]
                   + b1[c * NF + fp];
            }
        }
        Bt[(df * 64 + cl) * 72 + c] = (__bf16)vn;
    }
    __syncthreads();

    int lane = threadIdx.x & 31, wave = threadIdx.x >> 5;
    int g = lane >> 4, ln16 = lane & 15;
    for (int tt = wave; tt < 16; tt += 8) {
        int mt = tt >> 2, nt = tt & 3;
        v8f acc = {0.f, 0.f, 0.f, 0.f, 0.f, 0.f, 0.f, 0.f};
#pragma unroll
        for (int df = 0; df < 3; ++df) {
            const __bf16* arow = wcbf + df * 4096 + (mt * 16 + ln16) * 64;
            const __bf16* brow = Bt + (df * 64 + nt * 16 + ln16) * 72;
#pragma unroll
            for (int kc = 0; kc < 2; ++kc)
                acc = wmma_bf16(load_frag(arow + kc * 32 + g * 8),
                                load_frag(brow + kc * 32 + g * 8), acc);
        }
#pragma unroll
        for (int j = 0; j < 8; ++j) {
            int o = mt * 16 + g * 8 + j;
            int col = col0 + nt * 16 + ln16;
            if (col < NCOL)
                out[((size_t)(b * NC + o)) * NCOL + col] = acc[j] + bc[o];
        }
    }
}

// ---------------------------------------------------------------------------
// Host launcher.
// ---------------------------------------------------------------------------
extern "C" void kernel_launch(void* const* d_in, const int* in_sizes, int n_in,
                              void* d_out, int out_size, void* d_ws, size_t ws_size,
                              hipStream_t stream) {
    (void)in_sizes; (void)n_in; (void)out_size; (void)ws_size;

    const float* x          = (const float*)d_in[0];
    const float* log_dt     = (const float*)d_in[1];
    const float* log_A_real = (const float*)d_in[2];
    const float* A_imag     = (const float*)d_in[3];
    const float* C_re       = (const float*)d_in[4];
    const float* C_im       = (const float*)d_in[5];
    const float* Dskip      = (const float*)d_in[6];
    const float* Wout       = (const float*)d_in[7];
    const float* bout       = (const float*)d_in[8];
    const float* w0         = (const float*)d_in[9];
    const float* b0         = (const float*)d_in[10];
    const float* w1         = (const float*)d_in[11];
    const float* b1         = (const float*)d_in[12];
    const float* Wg         = (const float*)d_in[13];
    const float* bg         = (const float*)d_in[14];
    const float* Wi         = (const float*)d_in[15];
    const float* bi         = (const float*)d_in[16];
    const float* Wc         = (const float*)d_in[17];
    const float* bc         = (const float*)d_in[18];

    char* ws = (char*)d_ws;
    size_t off = 0;
    auto alloc = [&](size_t bytes) {
        size_t o = off;
        off += (bytes + 255) & ~(size_t)255;
        return o;
    };
    float*  kf     = (float*) (ws + alloc((size_t)PREP_NK * 4));
    __bf16* woutbf = (__bf16*)(ws + alloc((size_t)PREP_NWO * 2));
    __bf16* wgbf   = (__bf16*)(ws + alloc((size_t)PREP_NWG * 2));
    __bf16* wibf   = (__bf16*)(ws + alloc((size_t)PREP_NWG * 2));
    __bf16* wcbf   = (__bf16*)(ws + alloc((size_t)PREP_NWC * 2));
    float*  boutp  = (float*) (ws + alloc((size_t)OPAD * 4));
    __bf16* ubf    = (__bf16*)(ws + alloc((size_t)NBT * NF * NL * 2));
    float*  xr     = (float*) (ws + alloc((size_t)NB * NC * NCOL * 4));
    float*  hbuf   = (float*) (ws + alloc((size_t)NB * NC * NCOL * 4));
    float*  st0    = (float*) (ws + alloc((size_t)NBT * 2 * 4));
    float*  st1    = (float*) (ws + alloc((size_t)NBT * 2 * 4));

    // 1) S4D kernel K + bf16 weights + padded bias.
    prep_kernel<<<dim3((PREP_TOT + 255) / 256), dim3(256), 0, stream>>>(
        log_dt, log_A_real, A_imag, C_re, C_im, Wout, Wg, Wi, Wc, bout,
        kf, woutbf, wgbf, wibf, wcbf, boutp);

    // 2) Stage u[bt][f][c] bf16.
    stage_u_kernel<<<dim3(NB * NF * 4), dim3(256), 0, stream>>>(x, ubf);

    // 3) Fused S4D conv + GELU + Wout WMMA GEMM + GLU + residual.
    const size_t S4D_LDS = (size_t)OPAD * NL * 4 + (size_t)NL * KPAD * 2; // 110592
    s4d_glu_kernel<<<dim3(NBT), dim3(256), S4D_LDS, stream>>>(
        ubf, kf, Dskip, woutbf, boutp, x, xr);

    // 4) LN0 stats.
    ln_stats_kernel<<<dim3(NBT), dim3(256), 0, stream>>>(xr, st0);

    // 5) Gate: (Wi@x + bi) * sigmoid(Wg@LN0(x) + bg).
    dim3 gemmGrid((NCOL + 63) / 64, NB);
    gate_kernel<<<gemmGrid, dim3(256), 0, stream>>>(
        xr, st0, w0, b0, wgbf, bg, wibf, bi, hbuf);

    // 6) LN1 stats.
    ln_stats_kernel<<<dim3(NBT), dim3(256), 0, stream>>>(hbuf, st1);

    // 7) Final 3-tap conv over F with WMMA.
    outconv_kernel<<<gemmGrid, dim3(256), 0, stream>>>(
        hbuf, st1, w1, b1, wcbf, bc, (float*)d_out);
}
